// Decoder_38740605009937
// MI455X (gfx1250) — matrix operand order repaired and checked by reading, not compiled, not run on hardware
//
#include <hip/hip_runtime.h>
#include <hip/hip_bf16.h>

// Problem constants (match reference)
#define B_   64
#define T_   1024
#define L_   400
#define E_   512
#define D_   80
#define A_   512
#define K_   5
#define BT   16      // batch rows per workgroup (WMMA M dimension)
#define NTHR 512     // 16 waves of 32
#define NWAVE 16

typedef __attribute__((ext_vector_type(16))) __bf16 v16bf;
typedef __attribute__((ext_vector_type(4)))  __bf16 v4bf;
typedef __attribute__((ext_vector_type(8)))  float  v8f;

__device__ __forceinline__ __bf16 f2bf(float f) { return (__bf16)f; }
__device__ __forceinline__ float sigf(float x) { return 1.0f / (1.0f + expf(-x)); }
__device__ __forceinline__ float softplusf(float x) { return (x > 20.0f) ? x : log1pf(expf(x)); }

// ---- WMMA fragment loaders (CDNA5 ISA 7.12.2, wave32) ----
// A: 16x32 bf16 from LDS staging. lane&15 = row M; lane>>4 selects K-half.
__device__ __forceinline__ v16bf load_A_frag(const __bf16* As, int pitch, int k0, int lane) {
    int m = lane & 15, kh = lane >> 4;
    const __bf16* row = As + m * pitch + k0 + kh * 8;
    v16bf a;
#pragma unroll
    for (int e = 0; e < 16; ++e) a[e] = row[(e < 8) ? e : e + 8];
    return a;
}

// B: fragment-ordered packed weights; lane's 16 bf16 contiguous (one 32B load).
__device__ __forceinline__ v16bf load_B_fragP(const __bf16* Wf, int ntN, int kt, int nt, int lane) {
    const __bf16* p = Wf + ((((size_t)kt * ntN) + nt) * 32 + lane) * 16;
    return *(const v16bf*)p;
}

// Paired-tile GEMM: dst[16 x N] = A1 @ W1 (+ A2 @ W2) + biasP.
// Each wave owns pairs of 16-col tiles so one A fragment feeds two WMMAs.
// ACT: 0 = raw f32 -> dstF ; 1 = relu -> bf16 dstB ; 2 = tanh -> bf16 dstB.
template <int ACT>
__device__ __forceinline__ void gemm16P(const __bf16* A1, int p1, const __bf16* W1, int nkt1,
                                        const __bf16* A2, int p2, const __bf16* W2, int nkt2,
                                        const float* biasP, int N, int ntN,
                                        float* dstF, __bf16* dstB, int ldD,
                                        int wave, int lane) {
    int npair = ntN >> 1;   // Npad is a multiple of 32
    for (int p = wave; p < npair; p += NWAVE) {
        int nt0 = 2 * p, nt1 = nt0 + 1;
        int n = lane & 15;
        int col0 = (nt0 << 4) + n, col1 = col0 + 16;
        float bv0 = biasP[col0], bv1 = biasP[col1];
        v8f acc0, acc1;
#pragma unroll
        for (int i = 0; i < 8; ++i) { acc0[i] = bv0; acc1[i] = bv1; }
        for (int kt = 0; kt < nkt1; ++kt) {
            v16bf a  = load_A_frag(A1, p1, kt << 5, lane);
            v16bf b0 = load_B_fragP(W1, ntN, kt, nt0, lane);
            v16bf b1 = load_B_fragP(W1, ntN, kt, nt1, lane);
            acc0 = __builtin_amdgcn_wmma_f32_16x16x32_bf16(false, a, false, b0, (short)0, acc0, false, false);
            acc1 = __builtin_amdgcn_wmma_f32_16x16x32_bf16(false, a, false, b1, (short)0, acc1, false, false);
        }
        if (A2) {
            for (int kt = 0; kt < nkt2; ++kt) {
                v16bf a  = load_A_frag(A2, p2, kt << 5, lane);
                v16bf b0 = load_B_fragP(W2, ntN, kt, nt0, lane);
                v16bf b1 = load_B_fragP(W2, ntN, kt, nt1, lane);
                acc0 = __builtin_amdgcn_wmma_f32_16x16x32_bf16(false, a, false, b0, (short)0, acc0, false, false);
                acc1 = __builtin_amdgcn_wmma_f32_16x16x32_bf16(false, a, false, b1, (short)0, acc1, false, false);
            }
        }
        int mbase = (lane >> 4) * 8;   // D rows: m = i + 8*(lane>=16)
        if (ACT == 0) {
            if (col0 < N) {
#pragma unroll
                for (int i = 0; i < 8; ++i) dstF[(size_t)(mbase + i) * ldD + col0] = acc0[i];
            }
            if (col1 < N) {
#pragma unroll
                for (int i = 0; i < 8; ++i) dstF[(size_t)(mbase + i) * ldD + col1] = acc1[i];
            }
        } else {
#pragma unroll
            for (int i = 0; i < 8; ++i) {
                float v0 = acc0[i], v1 = acc1[i];
                if (ACT == 1) { v0 = fmaxf(v0, 0.0f); v1 = fmaxf(v1, 0.0f); }
                else          { v0 = tanhf(v0);        v1 = tanhf(v1); }
                dstB[(size_t)(mbase + i) * ldD + col0] = f2bf(v0);
                dstB[(size_t)(mbase + i) * ldD + col1] = f2bf(v1);
            }
        }
    }
}

// ---- prep kernels: pack f32 weights -> zero-padded, fragment-ordered bf16 ----
// rot: row rotation (k_src = (k + rot) % Kpad), used to permute lstm_Wx1 so the
// LDS A operand can be [h_att | ctx] instead of [ctx | h_att].
__global__ void prep_weight(const float* __restrict__ W, int K, int N,
                            __bf16* __restrict__ dst, int Kpad, int Npad, int rot) {
    int ntN = Npad >> 4;
    int total = Kpad * Npad;
    for (int idx = blockIdx.x * blockDim.x + threadIdx.x; idx < total; idx += gridDim.x * blockDim.x) {
        int e = idx & 15;
        int ln = (idx >> 4) & 31;
        int t  = idx >> 9;
        int nt = t % ntN;
        int kt = t / ntN;
        int k = (kt << 5) + ((ln >> 4) << 3) + ((e < 8) ? e : e + 8);
        int n = (nt << 4) + (ln & 15);
        int ks = (k + rot) % Kpad;
        float v = (ks < K && n < N) ? W[(size_t)ks * N + n] : 0.0f;
        dst[idx] = (__bf16)v;
    }
}

__global__ void prep_bias(const float* __restrict__ b, int N, float* __restrict__ dst, int Npad) {
    int i = blockIdx.x * blockDim.x + threadIdx.x;
    if (i < Npad) dst[i] = (i < N) ? b[i] : 0.0f;
}

__global__ void __launch_bounds__(NTHR)
decoder_scan_kernel(const float* __restrict__ enc_z,
                    const __bf16* pre_W1p, const float* pre_b1p,
                    const __bf16* pre_W2p, const float* pre_b2p,
                    const __bf16* gru_Wxp, const __bf16* gru_Whp,
                    const float* gru_bxp, const float* gru_bhp,
                    const __bf16* att_W1p, const float* att_b1p,
                    const __bf16* att_W2p, const float* att_b2p,
                    const __bf16* lstm_Wx1p, const __bf16* lstm_Wh1p, const float* lstm_b1p,
                    const __bf16* lstm_Wx2p, const __bf16* lstm_Wh2p, const float* lstm_b2p,
                    const float* stop_W, const float* stop_b,
                    float* out_after, float* out_before, float* out_logits)
{
    extern __shared__ char smem[];
    // LDS map (~316.9 KB of the 320 KB WGP budget)
    float*  h_att  = (float*)smem;                   // 16x512 f32 recurrent state
    float*  c1     = h_att + 16 * 512;               // 16x80
    float*  c2     = c1 + 16 * 80;
    float*  h2     = c2 + 16 * 80;
    float*  means  = h2 + 16 * 80;                   // 16x8 (K=5 padded)
    float*  scales = means + 16 * 8;
    float*  mixw   = scales + 16 * 8;
    float*  params = mixw + 16 * 8;                  // 16x16 (3K=15 padded)
    __bf16* AbfX   = (__bf16*)(params + 16 * 16);    // 16x1024: [h_att | ctx] bf16
    __bf16* QBf    = AbfX + 16 * 1024;               // 16x512 tanh(q) staging
    __bf16* Sprev  = QBf + 16 * 512;                 // 16x128 each (pitch 128):
    __bf16* S1     = Sprev + 16 * 128;
    __bf16* S2     = S1 + 16 * 128;
    __bf16* Sh1    = S2 + 16 * 128;
    __bf16* Sh2    = Sh1 + 16 * 128;
    float*  G1     = (float*)(Sh2 + 16 * 128);       // 16x1536 (gx | attn_w | lstm gates)
    float*  G2     = G1 + 16 * 1536;                 // 16x1536 (gh)
    float*  attn_w = G1;                             // 16x1024 alias

    const int tid  = threadIdx.x;
    const int lane = tid & 31;
    const int wave = tid >> 5;
    const int b0   = blockIdx.x * BT;

    // zero-init carry state + stagings (staging pads stay zero forever)
    for (int i = tid; i < 16 * 512; i += NTHR) h_att[i] = 0.0f;
    for (int i = tid; i < 16 * 80; i += NTHR) { c1[i] = 0; c2[i] = 0; h2[i] = 0; }
    for (int i = tid; i < 16 * 8; i += NTHR) means[i] = 0.0f;
    for (int i = tid; i < 16 * 1024; i += NTHR) AbfX[i] = f2bf(0.0f);
    for (int i = tid; i < 16 * 128; i += NTHR) { Sprev[i] = f2bf(0.0f); S1[i] = f2bf(0.0f); S2[i] = f2bf(0.0f); Sh1[i] = f2bf(0.0f); Sh2[i] = f2bf(0.0f); }
    __syncthreads();

    for (int step = 0; step < L_; ++step) {
        // --- overlap slot: gh = h@Wh+bh -> G2 (depends only on last step's AbfX h-staging)
        //     runs in the same barrier interval as prenet1 (only 3 pairs) so all
        //     16 waves have work.
        gemm16P<0>(AbfX, 1024, gru_Whp, 16, nullptr, 0, nullptr, 0, gru_bhp, 1536, 96, G2, nullptr, 1536, wave, lane);
        gemm16P<1>(Sprev, 128, pre_W1p, 3, nullptr, 0, nullptr, 0, pre_b1p, 80, 6, nullptr, S1, 128, wave, lane);
        __syncthreads();
        gemm16P<1>(S1, 128, pre_W2p, 3, nullptr, 0, nullptr, 0, pre_b2p, 80, 6, nullptr, S2, 128, wave, lane);
        __syncthreads();

        // --- GRU: gx = p@Wx+bx -> G1 ---
        gemm16P<0>(S2, 128, gru_Wxp, 3, nullptr, 0, nullptr, 0, gru_bxp, 1536, 96, G1, nullptr, 1536, wave, lane);
        __syncthreads();
        // GRU elementwise + fused bf16 re-staging of new h_att
        for (int i = tid; i < 16 * 512; i += NTHR) {
            int m = i >> 9, j = i & 511; int g = m * 1536 + j;
            float r = sigf(G1[g] + G2[g]);
            float z = sigf(G1[g + 512] + G2[g + 512]);
            float n = tanhf(G1[g + 1024] + r * G2[g + 1024]);
            float h = (1.0f - z) * n + z * h_att[i];
            h_att[i] = h;
            AbfX[m * 1024 + j] = f2bf(h);
        }
        __syncthreads();

        // --- MoL attention params: QBf = tanh(h@W1+b1) (fused); params = QBf@W2+b2 ---
        gemm16P<2>(AbfX, 1024, att_W1p, 16, nullptr, 0, nullptr, 0, att_b1p, 512, 32, nullptr, QBf, 512, wave, lane);
        __syncthreads();
        gemm16P<0>(QBf, 512, att_W2p, 16, nullptr, 0, nullptr, 0, att_b2p, 15, 2, params, nullptr, 16, wave, lane);
        __syncthreads();
        // mixture params: means/scales (threads 0..79), softmax (threads 128..143)
        if (tid < 16 * K_) {
            int m = tid / K_, k = tid % K_;
            means[m * 8 + k] += softplusf(params[m * 16 + k]);
            scales[m * 8 + k] = softplusf(params[m * 16 + 5 + k]) + 1e-4f;
        } else if (tid >= 128 && tid < 144) {
            int m = tid - 128;
            float mx = -1e30f;
            for (int k = 0; k < K_; ++k) mx = fmaxf(mx, params[m * 16 + 10 + k]);
            float s = 0.0f, e[K_];
            for (int k = 0; k < K_; ++k) { e[k] = expf(params[m * 16 + 10 + k] - mx); s += e[k]; }
            for (int k = 0; k < K_; ++k) mixw[m * 8 + k] = e[k] / s;
        }
        __syncthreads();

        // --- attention weights via logistic-CDF differences ---
        for (int i = tid; i < 16 * T_; i += NTHR) {
            int m = i >> 10, t = i & 1023;
            float pos = (float)t, acc = 0.0f;
#pragma unroll
            for (int k = 0; k < K_; ++k) {
                float mean = means[m * 8 + k], sc = scales[m * 8 + k];
                acc += mixw[m * 8 + k] * (sigf((pos + 0.5f - mean) / sc) - sigf((pos - 0.5f - mean) / sc));
            }
            attn_w[i] = acc;
        }
        __syncthreads();

        // --- context: wave w reduces batch row m=w over T (L2-resident enc_z stream);
        //     global_prefetch keeps the cache 8 rows ahead of the consume pointer;
        //     result written directly as bf16 into AbfX[:,512:1024] ---
        {
            int m = wave;
            const float* base = enc_z + (size_t)(b0 + m) * T_ * E_;
            float4 acc[4];
#pragma unroll
            for (int j = 0; j < 4; ++j) { acc[j].x = 0; acc[j].y = 0; acc[j].z = 0; acc[j].w = 0; }
            for (int t = 0; t < T_; ++t) {
                if (t + 8 < T_)   // 32 lanes x 64B = one full 2KB row per issue
                    __builtin_prefetch(base + (size_t)(t + 8) * E_ + lane * 16, 0, 3);
                float wv = attn_w[m * T_ + t];
                const float4* row = (const float4*)(base + (size_t)t * E_);
#pragma unroll
                for (int j = 0; j < 4; ++j) {
                    float4 v = row[j * 32 + lane];
                    acc[j].x += wv * v.x; acc[j].y += wv * v.y; acc[j].z += wv * v.z; acc[j].w += wv * v.w;
                }
            }
#pragma unroll
            for (int j = 0; j < 4; ++j) {
                v4bf pk; pk[0] = f2bf(acc[j].x); pk[1] = f2bf(acc[j].y); pk[2] = f2bf(acc[j].z); pk[3] = f2bf(acc[j].w);
                *(v4bf*)(AbfX + m * 1024 + 512 + 4 * (j * 32 + lane)) = pk;
            }
        }
        __syncthreads();

        // --- LSTM1: gates = [h_att|ctx]@rot(Wx1) + h1@Wh1 + b1 -> G1 ---
        gemm16P<0>(AbfX, 1024, lstm_Wx1p, 32, Sh1, 128, lstm_Wh1p, 3, lstm_b1p, 320, 20, G1, nullptr, 320, wave, lane);
        __syncthreads();
        for (int i = tid; i < 16 * 80; i += NTHR) {
            int m = i / 80, d = i % 80; int g = m * 320 + d;
            float ig = sigf(G1[g]), fg = sigf(G1[g + 80]), gg = tanhf(G1[g + 160]), og = sigf(G1[g + 240]);
            float c = fg * c1[i] + ig * gg;
            c1[i] = c;
            Sh1[m * 128 + d] = f2bf(og * tanhf(c));   // h1 lives only as bf16 staging
        }
        __syncthreads();

        // --- LSTM2: gates = h1@Wx2 + h2@Wh2 + b2 -> G1 ---
        gemm16P<0>(Sh1, 128, lstm_Wx2p, 3, Sh2, 128, lstm_Wh2p, 3, lstm_b2p, 320, 20, G1, nullptr, 320, wave, lane);
        __syncthreads();
        for (int i = tid; i < 16 * 80; i += NTHR) {
            int m = i / 80, d = i % 80; int g = m * 320 + d;
            float ig = sigf(G1[g]), fg = sigf(G1[g + 80]), gg = tanhf(G1[g + 160]), og = sigf(G1[g + 240]);
            float c = fg * c2[i] + ig * gg;
            c2[i] = c;
            float h = og * tanhf(c);
            h2[i] = h;
            __bf16 hb = f2bf(h);
            Sh2[m * 128 + d] = hb;                    // next-step Wh2 input
            Sprev[m * 128 + d] = hb;                  // next-step prenet input
            size_t o = ((size_t)(b0 + m) * L_ + step) * D_ + d;
            out_after[o] = h; out_before[o] = h;      // outputs fused here
        }
        __syncthreads();
        if (tid < 16) {
            int m = tid; float s = stop_b[0];
            for (int d = 0; d < 80; ++d) s += h2[m * 80 + d] * stop_W[d];
            out_logits[(size_t)(b0 + m) * L_ + step] = s;
        }
        __syncthreads();
    }
}

extern "C" void kernel_launch(void* const* d_in, const int* in_sizes, int n_in,
                              void* d_out, int out_size, void* d_ws, size_t ws_size,
                              hipStream_t stream) {
    const float* enc_z   = (const float*)d_in[0];
    // d_in[1] = ys: only its length (L) matters; values unused by the reference.
    const float* pre_W1  = (const float*)d_in[2];
    const float* pre_b1  = (const float*)d_in[3];
    const float* pre_W2  = (const float*)d_in[4];
    const float* pre_b2  = (const float*)d_in[5];
    const float* gru_Wx  = (const float*)d_in[6];
    const float* gru_Wh  = (const float*)d_in[7];
    const float* gru_bx  = (const float*)d_in[8];
    const float* gru_bh  = (const float*)d_in[9];
    const float* att_W1  = (const float*)d_in[10];
    const float* att_b1  = (const float*)d_in[11];
    const float* att_W2  = (const float*)d_in[12];
    const float* att_b2  = (const float*)d_in[13];
    const float* lstm_Wx1 = (const float*)d_in[14];
    const float* lstm_Wh1 = (const float*)d_in[15];
    const float* lstm_b1  = (const float*)d_in[16];
    const float* lstm_Wx2 = (const float*)d_in[17];
    const float* lstm_Wh2 = (const float*)d_in[18];
    const float* lstm_b2  = (const float*)d_in[19];
    const float* stop_W   = (const float*)d_in[20];
    const float* stop_b   = (const float*)d_in[21];

    // ---- workspace layout: packed bf16 weights (fragment order) + padded f32 biases ----
    __bf16* wsb = (__bf16*)d_ws;
    size_t o = 0;
    __bf16* pre_W1p   = wsb + o; o += 96 * 96;       // Kpad x Npad (Npad mult of 32)
    __bf16* pre_W2p   = wsb + o; o += 96 * 96;
    __bf16* gru_Wxp   = wsb + o; o += 96 * 1536;
    __bf16* gru_Whp   = wsb + o; o += 512 * 1536;
    __bf16* att_W1p   = wsb + o; o += 512 * 512;
    __bf16* att_W2p   = wsb + o; o += 512 * 32;
    __bf16* lstm_Wx1p = wsb + o; o += 1024 * 320;
    __bf16* lstm_Wh1p = wsb + o; o += 96 * 320;
    __bf16* lstm_Wx2p = wsb + o; o += 96 * 320;
    __bf16* lstm_Wh2p = wsb + o; o += 96 * 320;
    float* wsf = (float*)(wsb + o);
    size_t of = 0;
    float* pre_b1p  = wsf + of; of += 96;
    float* pre_b2p  = wsf + of; of += 96;
    float* gru_bxp  = wsf + of; of += 1536;
    float* gru_bhp  = wsf + of; of += 1536;
    float* att_b1p  = wsf + of; of += 512;
    float* att_b2p  = wsf + of; of += 32;
    float* lstm_b1p = wsf + of; of += 320;
    float* lstm_b2p = wsf + of; of += 320;

    struct WDesc { const float* src; int K, N; __bf16* dst; int Kp, Np, rot; };
    WDesc wd[10] = {
        { pre_W1,   80,   80,   pre_W1p,   96,   96,   0 },
        { pre_W2,   80,   80,   pre_W2p,   96,   96,   0 },
        { gru_Wx,   80,   1536, gru_Wxp,   96,   1536, 0 },
        { gru_Wh,   512,  1536, gru_Whp,   512,  1536, 0 },
        { att_W1,   512,  512,  att_W1p,   512,  512,  0 },
        { att_W2,   512,  15,   att_W2p,   512,  32,   0 },
        { lstm_Wx1, 1024, 320,  lstm_Wx1p, 1024, 320,  512 },  // A = [h_att|ctx]
        { lstm_Wh1, 80,   320,  lstm_Wh1p, 96,   320,  0 },
        { lstm_Wx2, 80,   320,  lstm_Wx2p, 96,   320,  0 },
        { lstm_Wh2, 80,   320,  lstm_Wh2p, 96,   320,  0 },
    };
    for (int i = 0; i < 10; ++i) {
        int total = wd[i].Kp * wd[i].Np;
        int blocks = (total + 255) / 256;
        if (blocks > 1024) blocks = 1024;
        prep_weight<<<dim3(blocks), dim3(256), 0, stream>>>(wd[i].src, wd[i].K, wd[i].N,
                                                            wd[i].dst, wd[i].Kp, wd[i].Np, wd[i].rot);
    }
    struct BDesc { const float* src; int N; float* dst; int Np; };
    BDesc bd[8] = {
        { pre_b1, 80, pre_b1p, 96 }, { pre_b2, 80, pre_b2p, 96 },
        { gru_bx, 1536, gru_bxp, 1536 }, { gru_bh, 1536, gru_bhp, 1536 },
        { att_b1, 512, att_b1p, 512 }, { att_b2, 15, att_b2p, 32 },
        { lstm_b1, 320, lstm_b1p, 320 }, { lstm_b2, 320, lstm_b2p, 320 },
    };
    for (int i = 0; i < 8; ++i) {
        int blocks = (bd[i].Np + 255) / 256;
        prep_bias<<<dim3(blocks), dim3(256), 0, stream>>>(bd[i].src, bd[i].N, bd[i].dst, bd[i].Np);
    }

    float* out = (float*)d_out;
    float* out_after  = out;
    float* out_before = out + (size_t)B_ * L_ * D_;
    float* out_logits = out + (size_t)2 * B_ * L_ * D_;

    // LDS: h_att 32768 + c1/c2/h2 15360 + mix 2560 + AbfX 32768 + QBf 16384
    //      + 5 stagings 20480 + G1 98304 + G2 98304 = 316928 B < 320 KB
    size_t shmem = 316928;

    decoder_scan_kernel<<<dim3(B_ / BT), dim3(NTHR), shmem, stream>>>(
        enc_z,
        pre_W1p, pre_b1p, pre_W2p, pre_b2p,
        gru_Wxp, gru_Whp, gru_bxp, gru_bhp,
        att_W1p, att_b1p, att_W2p, att_b2p,
        lstm_Wx1p, lstm_Wh1p, lstm_b1p,
        lstm_Wx2p, lstm_Wh2p, lstm_b2p,
        stop_W, stop_b,
        out_after, out_before, out_logits);
}
